// BANLayer_89893665505605
// MI455X (gfx1250) — matrix operand order repaired and checked by reading, not compiled, not run on hardware
//
#include <hip/hip_runtime.h>
#include <hip/hip_bf16.h>

// ---- CDNA5 WMMA types ------------------------------------------------------
typedef __attribute__((ext_vector_type(16))) _Float16 v16h;
typedef __attribute__((ext_vector_type(8)))  _Float16 v8h;
typedef __attribute__((ext_vector_type(8)))  float    v8f;

// A-matrix fragment (16x32 f16, M x K). ISA layout: lane L -> M = L%16;
// halves e<8 hold K = k0 + (L/16)*8 + e ; halves e>=8 hold K = +16.
__device__ __forceinline__ v16h frag_a_f32(const float* A, int lda, int row0,
                                           int k0, int lane) {
  const float* p = A + (size_t)(row0 + (lane & 15)) * lda + k0 + ((lane >> 4) << 3);
  v16h a;
#pragma unroll
  for (int t = 0; t < 8; ++t) {
    a[t]     = (_Float16)p[t];
    a[t + 8] = (_Float16)p[t + 16];
  }
  return a;
}

__device__ __forceinline__ v16h frag_a_f16(const _Float16* A, int lda, int row0,
                                           int k0, int lane) {
  const _Float16* p = A + (size_t)(row0 + (lane & 15)) * lda + k0 + ((lane >> 4) << 3);
  v8h lo = *(const v8h*)p;
  v8h hi = *(const v8h*)(p + 16);
  v16h a;
#pragma unroll
  for (int t = 0; t < 8; ++t) { a[t] = lo[t]; a[t + 8] = hi[t]; }
  return a;
}

// B-matrix fragment (32x16 f16, K x N). ISA layout: lane L -> N = col0 + L%16;
// halves e<8 hold K = k0 + (L/16)*8 + e ; halves e>=8 hold K = +16.
__device__ __forceinline__ v16h frag_b_f32(const float* B, int ldb, int k0,
                                           int col0, int lane) {
  const float* p = B + (size_t)(k0 + ((lane >> 4) << 3)) * ldb + col0 + (lane & 15);
  v16h b;
#pragma unroll
  for (int t = 0; t < 8; ++t) {
    b[t]     = (_Float16)p[(size_t)t * ldb];
    b[t + 8] = (_Float16)p[(size_t)(t + 16) * ldb];
  }
  return b;
}

__device__ __forceinline__ v16h frag_b_f16(const _Float16* B, int ldb, int k0,
                                           int col0, int lane) {
  const _Float16* p = B + (size_t)(k0 + ((lane >> 4) << 3)) * ldb + col0 + (lane & 15);
  v16h b;
#pragma unroll
  for (int t = 0; t < 8; ++t) {
    b[t]     = p[(size_t)t * ldb];
    b[t + 8] = p[(size_t)(t + 16) * ldb];
  }
  return b;
}

__device__ __forceinline__ v8f wmma_f16(v16h a, v16h b, v8f c) {
  return __builtin_amdgcn_wmma_f32_16x16x32_f16(false, a, false, b, (short)0, c,
                                                false, false);
}

// ---- constants -------------------------------------------------------------
#define NB    16
#define VNUM  1024
#define QNUM  512
#define VDIM  256
#define QDIM  128
#define NH    8
#define HD    64
#define PROJ  512   // NH*HD

// ---- zero the atomic accumulators (ws is poisoned by the harness) ----------
__global__ void ban_init(float* __restrict__ cbar, float* __restrict__ colsum) {
  int idx = blockIdx.x * blockDim.x + threadIdx.x;
  if (idx < NB * NH * QNUM) cbar[idx] = 0.f;
  if (idx < NB * PROJ) colsum[idx] = 0.f;
}

// ---- vp = v@Wv + bv, store f16 [B*V, 512]; also colsum over v per (b,col) --
__global__ void ban_proj_v(const float* __restrict__ v, const float* __restrict__ Wv,
                           const float* __restrict__ bv, _Float16* __restrict__ vp,
                           float* __restrict__ colsum) {
  const int lane = threadIdx.x & 31;
  const int wid  = blockIdx.x * (blockDim.x >> 5) + (threadIdx.x >> 5);
  const int row0 = (wid >> 4) * 32;        // M = 16384
  const int col0 = (wid & 15) * 32;        // N = 512
  v8f c00 = {}, c01 = {}, c10 = {}, c11 = {};
  for (int k0 = 0; k0 < VDIM; k0 += 32) {
    v16h a0 = frag_a_f32(v, VDIM, row0, k0, lane);
    v16h a1 = frag_a_f32(v, VDIM, row0 + 16, k0, lane);
    v16h b0 = frag_b_f32(Wv, PROJ, k0, col0, lane);
    v16h b1 = frag_b_f32(Wv, PROJ, k0, col0 + 16, lane);
    c00 = wmma_f16(a0, b0, c00);
    c01 = wmma_f16(a0, b1, c01);
    c10 = wmma_f16(a1, b0, c10);
    c11 = wmma_f16(a1, b1, c11);
  }
  const int b = row0 >> 10;                // 1024 rows per batch, tiles never cross
  const int half8 = (lane >> 4) << 3;
  v8f acc[4] = {c00, c01, c10, c11};
#pragma unroll
  for (int i = 0; i < 2; ++i)
#pragma unroll
    for (int j = 0; j < 2; ++j) {
      v8f c = acc[i * 2 + j];
      int col = col0 + 16 * j + (lane & 15);
      float bias = bv[col];
      float cs = 0.f;
#pragma unroll
      for (int e = 0; e < 8; ++e) {
        float val = c[e] + bias;
        int row = row0 + 16 * i + half8 + e;
        vp[(size_t)row * PROJ + col] = (_Float16)val;
        cs += val;
      }
      atomicAdd(&colsum[b * PROJ + col], cs);
    }
}

// ---- qp = q@Wq + bq, stored TRANSPOSED per head: qpt[b][h][k(64)][q(512)] ---
__global__ void ban_proj_q(const float* __restrict__ q, const float* __restrict__ Wq,
                           const float* __restrict__ bq, _Float16* __restrict__ qpt) {
  const int lane = threadIdx.x & 31;
  const int wid  = blockIdx.x * (blockDim.x >> 5) + (threadIdx.x >> 5);
  const int row0 = (wid >> 4) * 32;        // M = 8192
  const int col0 = (wid & 15) * 32;
  v8f c00 = {}, c01 = {}, c10 = {}, c11 = {};
  for (int k0 = 0; k0 < QDIM; k0 += 32) {
    v16h a0 = frag_a_f32(q, QDIM, row0, k0, lane);
    v16h a1 = frag_a_f32(q, QDIM, row0 + 16, k0, lane);
    v16h b0 = frag_b_f32(Wq, PROJ, k0, col0, lane);
    v16h b1 = frag_b_f32(Wq, PROJ, k0, col0 + 16, lane);
    c00 = wmma_f16(a0, b0, c00);
    c01 = wmma_f16(a0, b1, c01);
    c10 = wmma_f16(a1, b0, c10);
    c11 = wmma_f16(a1, b1, c11);
  }
  const int b = row0 >> 9;                 // 512 rows per batch
  const int half8 = (lane >> 4) << 3;
  v8f acc[4] = {c00, c01, c10, c11};
#pragma unroll
  for (int i = 0; i < 2; ++i)
#pragma unroll
    for (int j = 0; j < 2; ++j) {
      v8f c = acc[i * 2 + j];
      int col = col0 + 16 * j + (lane & 15);
      int h = col >> 6, k = col & 63;
      float bias = bq[col];
      int qrow = (row0 & (QNUM - 1)) + 16 * i + half8;   // 8-aligned
      v8h outv;
#pragma unroll
      for (int e = 0; e < 8; ++e) outv[e] = (_Float16)(c[e] + bias);
      *(v8h*)(qpt + ((size_t)((b * NH + h) * HD + k)) * QNUM + qrow) = outv;
    }
}

// ---- Kt[b][h] = vp_head[1024x64] @ att_w[h][64x64] * scale, f16 ------------
__global__ void ban_kt(const _Float16* __restrict__ vp, const float* __restrict__ attw,
                       _Float16* __restrict__ Kt) {
  const int lane = threadIdx.x & 31;
  const int wid  = blockIdx.x * (blockDim.x >> 5) + (threadIdx.x >> 5);
  const int bh   = wid >> 6;               // 128 (b,h) pairs
  const int t    = wid & 63;               // 32 row-tiles x 2 col-tiles
  const int b = bh >> 3, h = bh & 7;
  const int vrow0 = (t >> 1) * 32;
  const int col0  = (t & 1) * 32;
  const int grow0 = b * VNUM + vrow0;
  const float* Ah = attw + (size_t)h * HD * HD;
  v8f c00 = {}, c01 = {}, c10 = {}, c11 = {};
  for (int k0 = 0; k0 < HD; k0 += 32) {
    v16h a0 = frag_a_f16(vp, PROJ, grow0, h * HD + k0, lane);
    v16h a1 = frag_a_f16(vp, PROJ, grow0 + 16, h * HD + k0, lane);
    v16h b0 = frag_b_f32(Ah, HD, k0, col0, lane);
    v16h b1 = frag_b_f32(Ah, HD, k0, col0 + 16, lane);
    c00 = wmma_f16(a0, b0, c00);
    c01 = wmma_f16(a0, b1, c01);
    c10 = wmma_f16(a1, b0, c10);
    c11 = wmma_f16(a1, b1, c11);
  }
  const float scale = 0.125f;              // hd^-0.5, folded into Kt
  const int half8 = (lane >> 4) << 3;
  v8f acc[4] = {c00, c01, c10, c11};
#pragma unroll
  for (int i = 0; i < 2; ++i)
#pragma unroll
    for (int j = 0; j < 2; ++j) {
      v8f c = acc[i * 2 + j];
      int col = col0 + 16 * j + (lane & 15);
#pragma unroll
      for (int e = 0; e < 8; ++e) {
        int vr = vrow0 + 16 * i + half8 + e;
        Kt[((size_t)bh * VNUM + vr) * HD + col] = (_Float16)(c[e] * scale);
      }
    }
}

// ---- logits(16x512 per WG) -> softmax -> colsum accumulation (fused) -------
// block: 256 thr (8 waves). Wave w computes cols [w*64, w*64+64). Logits never
// touch global memory; only cbar[b,h,q] = sum_v softmax_w is written.
__global__ void ban_logits(const _Float16* __restrict__ Kt,
                           const _Float16* __restrict__ qpt,
                           float* __restrict__ cbar) {
  const int bh   = blockIdx.x >> 6;        // 128
  const int row0 = (blockIdx.x & 63) * 16; // 64 row-blocks of 16
  const int lane = threadIdx.x & 31;
  const int wave = threadIdx.x >> 5;
  const int col0 = wave * 64;
  const int hi   = lane >> 4;              // half: rows 0-7 vs 8-15
  const _Float16* Kb = Kt + (size_t)bh * VNUM * HD;
  const _Float16* Qb = qpt + (size_t)bh * HD * QNUM;

  v8f c[4] = {v8f{}, v8f{}, v8f{}, v8f{}};
  for (int k0 = 0; k0 < HD; k0 += 32) {
    v16h a = frag_a_f16(Kb, HD, row0, k0, lane);
#pragma unroll
    for (int t = 0; t < 4; ++t) {
      v16h bf = frag_b_f16(Qb, QNUM, k0, col0 + 16 * t, lane);
      c[t] = wmma_f16(a, bf, c[t]);
    }
  }

  __shared__ float red[8][16];
  __shared__ float rowstat[16];

  // 1) row max: per-lane over 4 col-tiles, butterfly across 16 lanes
  float m[8];
#pragma unroll
  for (int e = 0; e < 8; ++e)
    m[e] = fmaxf(fmaxf(c[0][e], c[1][e]), fmaxf(c[2][e], c[3][e]));
#pragma unroll
  for (int off = 1; off < 16; off <<= 1)
#pragma unroll
    for (int e = 0; e < 8; ++e) m[e] = fmaxf(m[e], __shfl_xor(m[e], off, 32));
  if ((lane & 15) == 0)
#pragma unroll
    for (int e = 0; e < 8; ++e) red[wave][hi * 8 + e] = m[e];
  __syncthreads();
  if (threadIdx.x < 16) {
    float v = red[0][threadIdx.x];
#pragma unroll
    for (int w = 1; w < 8; ++w) v = fmaxf(v, red[w][threadIdx.x]);
    rowstat[threadIdx.x] = v;
  }
  __syncthreads();
  float rmax[8];
#pragma unroll
  for (int e = 0; e < 8; ++e) rmax[e] = rowstat[hi * 8 + e];

  // 2) exp + row sum
  float s[8] = {0, 0, 0, 0, 0, 0, 0, 0};
#pragma unroll
  for (int t = 0; t < 4; ++t)
#pragma unroll
    for (int e = 0; e < 8; ++e) {
      float ev = __expf(c[t][e] - rmax[e]);
      c[t][e] = ev;
      s[e] += ev;
    }
#pragma unroll
  for (int off = 1; off < 16; off <<= 1)
#pragma unroll
    for (int e = 0; e < 8; ++e) s[e] += __shfl_xor(s[e], off, 32);
  if ((lane & 15) == 0)
#pragma unroll
    for (int e = 0; e < 8; ++e) red[wave][hi * 8 + e] = s[e];
  __syncthreads();
  if (threadIdx.x < 16) {
    float v = 0.f;
#pragma unroll
    for (int w = 0; w < 8; ++w) v += red[w][threadIdx.x];
    rowstat[threadIdx.x] = v;
  }
  __syncthreads();
  float rinv[8];
#pragma unroll
  for (int e = 0; e < 8; ++e) rinv[e] = 1.0f / rowstat[hi * 8 + e];

  // 3) normalized column sums over these 16 rows -> cbar
#pragma unroll
  for (int t = 0; t < 4; ++t) {
    float cs = 0.f;
#pragma unroll
    for (int e = 0; e < 8; ++e) cs += c[t][e] * rinv[e];
    cs += __shfl_xor(cs, 16, 32);          // combine the two row-halves
    if (hi == 0)
      atomicAdd(&cbar[(size_t)bh * QNUM + col0 + 16 * t + (lane & 15)], cs);
  }
}

// ---- pooled_v / pooled_q -> fused[b, 1024] (exact f32 tail) ----------------
__global__ void ban_pool(const _Float16* __restrict__ qpt, const float* __restrict__ cbar,
                         const float* __restrict__ colsum, float* __restrict__ fused) {
  int idx = blockIdx.x * blockDim.x + threadIdx.x;   // 16*1024
  int b = idx >> 10, r = idx & 1023;
  int h = r >> 7, jj = r & 127;
  float val;
  if (jj < HD) {
    const float* cb = cbar + (size_t)(b * NH + h) * QNUM;
    const _Float16* qr = qpt + ((size_t)(b * NH + h) * HD + jj) * QNUM;
    float sum = 0.f;
    for (int qi = 0; qi < QNUM; ++qi) sum += cb[qi] * (float)qr[qi];
    val = sum * (1.0f / (float)VNUM);                // pooled_v
  } else {
    // softmax rows sum to 1 => pooled_q = (1/Q) * colsum_v(vp)
    val = colsum[b * PROJ + h * HD + (jj - HD)] * (1.0f / (float)QNUM);
  }
  fused[idx] = val;
}

// ---- out = relu(fused @ Wo + bo), 16x1024x512 (f32) ------------------------
__global__ void ban_out(const float* __restrict__ fused, const float* __restrict__ Wo,
                        const float* __restrict__ bo, float* __restrict__ out) {
  int idx = blockIdx.x * blockDim.x + threadIdx.x;   // 16*512
  int b = idx >> 9, o = idx & 511;
  float acc = bo[o];
  const float* f = fused + (size_t)b * 1024;
  for (int k = 0; k < 1024; ++k) acc = fmaf(f[k], Wo[(size_t)k * 512 + o], acc);
  out[idx] = fmaxf(acc, 0.f);
}

// ---- launcher --------------------------------------------------------------
extern "C" void kernel_launch(void* const* d_in, const int* in_sizes, int n_in,
                              void* d_out, int out_size, void* d_ws, size_t ws_size,
                              hipStream_t stream) {
  (void)in_sizes; (void)n_in; (void)out_size; (void)ws_size;
  const float* v    = (const float*)d_in[0];
  const float* q    = (const float*)d_in[1];
  const float* Wv   = (const float*)d_in[2];
  const float* bv   = (const float*)d_in[3];
  const float* Wq   = (const float*)d_in[4];
  const float* bq   = (const float*)d_in[5];
  const float* attw = (const float*)d_in[6];
  const float* Wo   = (const float*)d_in[7];
  const float* bo   = (const float*)d_in[8];
  float* out = (float*)d_out;

  char* ws = (char*)d_ws;
  _Float16* vp  = (_Float16*)(ws);                       // 16 MiB  [B*V, 512] f16
  _Float16* qpt = (_Float16*)(ws + 16777216);            //  8 MiB  [B,H,64,512] f16
  _Float16* Kt  = (_Float16*)(ws + 25165824);            // 16 MiB  [B,H,1024,64] f16
  float* cbar   = (float*)(ws + 41943040);               // 256 KiB [B,H,512] f32
  float* colsum = (float*)(ws + 42205184);               //  32 KiB [B,512] f32
  float* fused  = (float*)(ws + 42237952);               //  64 KiB [B,1024] f32

  ban_init  <<<256,  256, 0, stream>>>(cbar, colsum);
  ban_proj_v<<<2048, 128, 0, stream>>>(v, Wv, bv, vp, colsum);
  ban_proj_q<<<1024, 128, 0, stream>>>(q, Wq, bq, qpt);
  ban_kt    <<<2048, 128, 0, stream>>>(vp, attw, Kt);
  ban_logits<<<8192, 256, 0, stream>>>(Kt, qpt, cbar);
  ban_pool  <<<64,   256, 0, stream>>>(qpt, cbar, colsum, fused);
  ban_out   <<<32,   256, 0, stream>>>(fused, Wo, bo, out);
}
